// CachedSpectralGPSLayer_29635274342671
// MI455X (gfx1250) — matrix operand order repaired and checked by reading, not compiled, not run on hardware
//
#include <hip/hip_runtime.h>

typedef __attribute__((ext_vector_type(16))) __bf16 v16bf;
typedef __attribute__((ext_vector_type(8)))  float  v8f;

static constexpr int Nn   = 32768;
static constexpr int Cc   = 128;
static constexpr int Ee   = 524288;
static constexpr int KEIG = 128;
static constexpr int Bg   = 64;
static constexpr int NG   = 512;
static constexpr int Hh   = 4;
static constexpr int KCH  = 5;

__device__ __forceinline__ unsigned short f2bfu(float f) {
    unsigned int u = __builtin_bit_cast(unsigned int, f);
    return (unsigned short)(u >> 16);
}
__device__ __forceinline__ v8f v8f_zero() {
    v8f z; for (int i = 0; i < 8; ++i) z[i] = 0.0f; return z;
}

struct Raw32 { uint4 lo, hi; };

// 16 bf16 from two 16B-aligned LDS chunks -> two ds_load_b128
__device__ __forceinline__ v16bf frag_ld2(const unsigned short* p0, const unsigned short* p1) {
    Raw32 t;
    t.lo = *(const uint4*)p0;
    t.hi = *(const uint4*)p1;
    return __builtin_bit_cast(v16bf, t);
}
// 16 contiguous bf16 (32B, 16B-aligned)
__device__ __forceinline__ v16bf frag_ld(const unsigned short* p) {
    return frag_ld2(p, p + 8);
}
// pack float4 -> 4 bf16 -> single 8B LDS store
__device__ __forceinline__ void st4bf(unsigned short* p, float4 v) {
    unsigned long long q = (unsigned long long)f2bfu(v.x)
        | ((unsigned long long)f2bfu(v.y) << 16)
        | ((unsigned long long)f2bfu(v.z) << 32)
        | ((unsigned long long)f2bfu(v.w) << 48);
    *(unsigned long long*)p = q;
}

// ---------------------------------------------------------------------------
// Generic bf16 WMMA GEMM: out[M,Nc] = op(A[M,K] @ W[K,Nc] + bias (+resid)).
// Block tile 128x64, 8 waves, each wave 32x32 (2x2 of 16x16 WMMA tiles).
// Double-buffered LDS ping-pong: stage chunk t+1 before computing chunk t so
// global-load latency overlaps the WMMA pipe. One barrier per iteration.
// ---------------------------------------------------------------------------
__launch_bounds__(256)
__global__ void k_gemm(const float* __restrict__ A, const float* __restrict__ W,
                       const float* __restrict__ bias, const float* __restrict__ resid,
                       float* __restrict__ out, int M, int K, int Nc, int relu)
{
    __shared__ unsigned short As[2][128][40];   // [buf][m][k], 80B rows (16B aligned)
    __shared__ unsigned short WsT[2][64][40];   // [buf][col][k] (transposed)

    const int tid  = threadIdx.x;
    const int lane = tid & 31;
    const int wave = tid >> 5;
    const int wm   = wave & 3;
    const int wn   = wave >> 2;
    const int bm   = blockIdx.y * 128;
    const int bn   = blockIdx.x * 64;

    v8f acc[2][2];
    for (int i = 0; i < 2; ++i) for (int j = 0; j < 2; ++j) acc[i][j] = v8f_zero();

    const int arow = lane & 15;
    const int akb  = (lane >> 4) ? 8 : 0;    // A-frag K base (ISA 16-bit A layout)
    const int bcol = lane & 15;
    const int bkb  = (lane >> 4) ? 8 : 0;    // B-frag K base

    auto stage = [&](int buf, int k0) {
        #pragma unroll
        for (int i = tid; i < 1024; i += 256) {          // A: 128x32 as float4
            int r = i >> 3, c4 = (i & 7) << 2;
            float4 av = *(const float4*)(A + (size_t)(bm + r) * K + k0 + c4);
            st4bf(&As[buf][r][c4], av);
        }
        #pragma unroll
        for (int i = tid; i < 512; i += 256) {           // W: 32x64 as float4, transpose
            int r = i >> 4, c4 = (i & 15) << 2;
            float4 wv = *(const float4*)(W + (size_t)(k0 + r) * Nc + bn + c4);
            WsT[buf][c4 + 0][r] = f2bfu(wv.x);
            WsT[buf][c4 + 1][r] = f2bfu(wv.y);
            WsT[buf][c4 + 2][r] = f2bfu(wv.z);
            WsT[buf][c4 + 3][r] = f2bfu(wv.w);
        }
    };

    const int nk = K >> 5;
    stage(0, 0);
    __syncthreads();

    for (int t = 0; t < nk; ++t) {
        const int buf = t & 1;
        if (t + 1 < nk) stage(buf ^ 1, (t + 1) << 5);    // prefetch next chunk

        const unsigned short* a0 = &As[buf][wm * 32 + arow][0];
        const unsigned short* a1 = &As[buf][wm * 32 + 16 + arow][0];
        v16bf af0 = frag_ld2(a0 + akb, a0 + akb + 16);
        v16bf af1 = frag_ld2(a1 + akb, a1 + akb + 16);
        v16bf bf0 = frag_ld2(&WsT[buf][wn * 32 + bcol][bkb], &WsT[buf][wn * 32 + bcol][bkb + 16]);
        v16bf bf1 = frag_ld2(&WsT[buf][wn * 32 + 16 + bcol][bkb], &WsT[buf][wn * 32 + 16 + bcol][bkb + 16]);

        acc[0][0] = __builtin_amdgcn_wmma_f32_16x16x32_bf16(false, af0, false, bf0, (short)0, acc[0][0], false, false);
        acc[0][1] = __builtin_amdgcn_wmma_f32_16x16x32_bf16(false, af0, false, bf1, (short)0, acc[0][1], false, false);
        acc[1][0] = __builtin_amdgcn_wmma_f32_16x16x32_bf16(false, af1, false, bf0, (short)0, acc[1][0], false, false);
        acc[1][1] = __builtin_amdgcn_wmma_f32_16x16x32_bf16(false, af1, false, bf1, (short)0, acc[1][1], false, false);
        __syncthreads();
    }

    const int mbase = (lane >> 4) ? 8 : 0;   // D layout: vgpr r -> M = r + mbase
    const bool has_resid = resid != nullptr;
    #pragma unroll
    for (int i = 0; i < 2; ++i) {
        #pragma unroll
        for (int j = 0; j < 2; ++j) {
            const int col  = bn + wn * 32 + j * 16 + bcol;
            const int row0 = bm + wm * 32 + i * 16 + mbase;
            const float bv = bias ? bias[col] : 0.0f;
            float rv[8];
            if (has_resid) {
                #pragma unroll
                for (int r = 0; r < 8; ++r)
                    rv[r] = resid[(size_t)(row0 + r) * Nc + col];
            }
            #pragma unroll
            for (int r = 0; r < 8; ++r) {
                float v = acc[i][j][r] + bv;
                if (has_resid) v += rv[r];
                if (relu) v = fmaxf(v, 0.0f);
                out[(size_t)(row0 + r) * Nc + col] = v;
            }
        }
    }
}

// ---------------------------------------------------------------------------
// Split-K transposed-A GEMM: out[M,Nc] += A[K,M]^T @ W[K,Nc]  (atomic f32).
// Used for U^T @ h_proj (M=KEIG=128, K=N=32768). out must be pre-zeroed.
// Same double-buffered pipeline (16 chunks per block -> overlap matters most).
// ---------------------------------------------------------------------------
__launch_bounds__(256)
__global__ void k_gemm_atb(const float* __restrict__ A, const float* __restrict__ W,
                           float* __restrict__ out, int M, int K, int Nc, int kchunk)
{
    __shared__ unsigned short As[2][128][40];
    __shared__ unsigned short WsT[2][64][40];

    const int tid  = threadIdx.x;
    const int lane = tid & 31;
    const int wave = tid >> 5;
    const int wm   = wave & 3;
    const int wn   = wave >> 2;
    const int bm   = blockIdx.y * 128;
    const int bn   = blockIdx.x * 64;
    const int kb0  = blockIdx.z * kchunk;

    v8f acc[2][2];
    for (int i = 0; i < 2; ++i) for (int j = 0; j < 2; ++j) acc[i][j] = v8f_zero();

    const int arow = lane & 15;
    const int akb  = (lane >> 4) ? 8 : 0;
    const int bcol = lane & 15;
    const int bkb  = (lane >> 4) ? 8 : 0;

    auto stage = [&](int buf, int k0) {
        #pragma unroll
        for (int i = tid; i < 1024; i += 256) {          // A[k][m] -> As[m][k]
            int c = i >> 5, m4 = (i & 31) << 2;          // coalesced float4 over m
            float4 av = *(const float4*)(A + (size_t)(k0 + c) * M + bm + m4);
            As[buf][m4 + 0][c] = f2bfu(av.x);
            As[buf][m4 + 1][c] = f2bfu(av.y);
            As[buf][m4 + 2][c] = f2bfu(av.z);
            As[buf][m4 + 3][c] = f2bfu(av.w);
        }
        #pragma unroll
        for (int i = tid; i < 512; i += 256) {
            int r = i >> 4, c4 = (i & 15) << 2;
            float4 wv = *(const float4*)(W + (size_t)(k0 + r) * Nc + bn + c4);
            WsT[buf][c4 + 0][r] = f2bfu(wv.x);
            WsT[buf][c4 + 1][r] = f2bfu(wv.y);
            WsT[buf][c4 + 2][r] = f2bfu(wv.z);
            WsT[buf][c4 + 3][r] = f2bfu(wv.w);
        }
    };

    const int nk = kchunk >> 5;
    stage(0, kb0);
    __syncthreads();

    for (int t = 0; t < nk; ++t) {
        const int buf = t & 1;
        if (t + 1 < nk) stage(buf ^ 1, kb0 + ((t + 1) << 5));

        const unsigned short* a0 = &As[buf][wm * 32 + arow][0];
        const unsigned short* a1 = &As[buf][wm * 32 + 16 + arow][0];
        v16bf af0 = frag_ld2(a0 + akb, a0 + akb + 16);
        v16bf af1 = frag_ld2(a1 + akb, a1 + akb + 16);
        v16bf bf0 = frag_ld2(&WsT[buf][wn * 32 + bcol][bkb], &WsT[buf][wn * 32 + bcol][bkb + 16]);
        v16bf bf1 = frag_ld2(&WsT[buf][wn * 32 + 16 + bcol][bkb], &WsT[buf][wn * 32 + 16 + bcol][bkb + 16]);

        acc[0][0] = __builtin_amdgcn_wmma_f32_16x16x32_bf16(false, af0, false, bf0, (short)0, acc[0][0], false, false);
        acc[0][1] = __builtin_amdgcn_wmma_f32_16x16x32_bf16(false, af0, false, bf1, (short)0, acc[0][1], false, false);
        acc[1][0] = __builtin_amdgcn_wmma_f32_16x16x32_bf16(false, af1, false, bf0, (short)0, acc[1][0], false, false);
        acc[1][1] = __builtin_amdgcn_wmma_f32_16x16x32_bf16(false, af1, false, bf1, (short)0, acc[1][1], false, false);
        __syncthreads();
    }

    const int mbase = (lane >> 4) ? 8 : 0;
    #pragma unroll
    for (int i = 0; i < 2; ++i)
        #pragma unroll
        for (int j = 0; j < 2; ++j) {
            int col = bn + wn * 32 + j * 16 + bcol;
            #pragma unroll
            for (int r = 0; r < 8; ++r) {
                int row = bm + wm * 32 + i * 16 + mbase + r;
                atomicAdd(&out[(size_t)row * Nc + col], acc[i][j][r]);
            }
        }
}

// ---------------------------------------------------------------------------
// Flash attention, dh=32, B=64 graphs of n=512, H=4 heads.
// Block = 4 waves; each wave owns a 16-row query tile. K staged [key][d]
// (contiguous d per key -> contiguous S-B-fragments), V staged transposed
// [d][key] (contiguous key per d -> contiguous V-B-fragments), 256-key chunks.
// ---------------------------------------------------------------------------
__launch_bounds__(128)
__global__ void k_attn(const float* __restrict__ qkv, float* __restrict__ o)
{
    __shared__ unsigned short Ks[256][32];    // [key][d], 64B rows
    __shared__ unsigned short VsT[32][264];   // [d][key], 528B rows (16B aligned)
    __shared__ unsigned short Ps[4][16][40];  // per-wave P staging, 80B rows

    const int tid  = threadIdx.x;
    const int lane = tid & 31;
    const int wave = tid >> 5;
    const int bh   = blockIdx.x;
    const int gb   = bh / Hh;
    const int h    = bh % Hh;
    const size_t rowbase = (size_t)gb * NG;

    const int arow  = lane & 15;
    const int akb   = (lane >> 4) ? 8 : 0;
    const int bcol  = lane & 15;
    const int bkb   = (lane >> 4) ? 8 : 0;
    const int mbase = (lane >> 4) ? 8 : 0;

    // Q fragment (16x32) for this wave's rows, from global fp32 (once).
    const int q0 = blockIdx.y * 64 + wave * 16;
    v16bf qf;
    #pragma unroll
    for (int v = 0; v < 8; ++v) {
        int kk = ((v < 4) ? (2 * v) : (16 + 2 * (v - 4))) + akb;
        const float2 qp = *(const float2*)&qkv[(rowbase + q0 + arow) * 384 + h * 32 + kk];
        qf[2 * v]     = __builtin_bit_cast(__bf16, f2bfu(qp.x));
        qf[2 * v + 1] = __builtin_bit_cast(__bf16, f2bfu(qp.y));
    }

    float mrow[8], lrow[8];
    for (int r = 0; r < 8; ++r) { mrow[r] = -1e30f; lrow[r] = 0.0f; }
    v8f o0 = v8f_zero(), o1 = v8f_zero();
    const float sscale = 0.17677669529663687f; // 1/sqrt(32)

    for (int c0 = 0; c0 < NG; c0 += 256) {
        __syncthreads();
        #pragma unroll
        for (int i = tid; i < 2048; i += 128) {   // 256 keys x 32 dims, float4
            int r = i >> 3, c4 = (i & 7) << 2;
            size_t grow = (rowbase + c0 + r) * 384 + h * 32;
            float4 kv = *(const float4*)&qkv[grow + 128 + c4];
            float4 vv = *(const float4*)&qkv[grow + 256 + c4];
            st4bf(&Ks[r][c4], kv);
            VsT[c4 + 0][r] = f2bfu(vv.x);
            VsT[c4 + 1][r] = f2bfu(vv.y);
            VsT[c4 + 2][r] = f2bfu(vv.z);
            VsT[c4 + 3][r] = f2bfu(vv.w);
        }
        __syncthreads();

        for (int j0 = 0; j0 < 256; j0 += 32) {
            // S = Q @ K^T for 32 keys (two 16-col tiles)
            v16bf bf0 = frag_ld2(&Ks[j0 + bcol][bkb], &Ks[j0 + bcol][bkb + 16]);
            v16bf bf1 = frag_ld2(&Ks[j0 + 16 + bcol][bkb], &Ks[j0 + 16 + bcol][bkb + 16]);
            v8f s0 = __builtin_amdgcn_wmma_f32_16x16x32_bf16(
                false, qf, false, bf0, (short)0, v8f_zero(), false, false);
            v8f s1 = __builtin_amdgcn_wmma_f32_16x16x32_bf16(
                false, qf, false, bf1, (short)0, v8f_zero(), false, false);

            float rmax[8];
            #pragma unroll
            for (int r = 0; r < 8; ++r) {
                s0[r] *= sscale; s1[r] *= sscale;
                rmax[r] = fmaxf(s0[r], s1[r]);
            }
            #pragma unroll
            for (int m = 1; m < 16; m <<= 1)
                #pragma unroll
                for (int r = 0; r < 8; ++r)
                    rmax[r] = fmaxf(rmax[r], __shfl_xor(rmax[r], m, 32));

            float alpha[8], rs[8];
            #pragma unroll
            for (int r = 0; r < 8; ++r) {
                float mn = fmaxf(mrow[r], rmax[r]);
                alpha[r] = __expf(mrow[r] - mn);
                mrow[r] = mn;
                s0[r] = __expf(s0[r] - mn);
                s1[r] = __expf(s1[r] - mn);
                rs[r] = s0[r] + s1[r];
            }
            #pragma unroll
            for (int m = 1; m < 16; m <<= 1)
                #pragma unroll
                for (int r = 0; r < 8; ++r)
                    rs[r] += __shfl_xor(rs[r], m, 32);
            #pragma unroll
            for (int r = 0; r < 8; ++r) {
                lrow[r] = lrow[r] * alpha[r] + rs[r];
                o0[r] *= alpha[r];
                o1[r] *= alpha[r];
            }

            // Re-layout P (D-layout) -> A-fragment via per-wave LDS tile.
            #pragma unroll
            for (int r = 0; r < 8; ++r) {
                Ps[wave][mbase + r][bcol]      = f2bfu(s0[r]);
                Ps[wave][mbase + r][16 + bcol] = f2bfu(s1[r]);
            }
            const unsigned short* pp = &Ps[wave][arow][0];
            v16bf pf = frag_ld2(pp + akb, pp + akb + 16);

            // O += P @ V (two d-halves), V fragments contiguous from VsT.
            v16bf vb0 = frag_ld2(&VsT[bcol][j0 + bkb], &VsT[bcol][j0 + bkb + 16]);
            v16bf vb1 = frag_ld2(&VsT[16 + bcol][j0 + bkb], &VsT[16 + bcol][j0 + bkb + 16]);
            o0 = __builtin_amdgcn_wmma_f32_16x16x32_bf16(
                false, pf, false, vb0, (short)0, o0, false, false);
            o1 = __builtin_amdgcn_wmma_f32_16x16x32_bf16(
                false, pf, false, vb1, (short)0, o1, false, false);
        }
    }

    #pragma unroll
    for (int r = 0; r < 8; ++r) {
        float inv = 1.0f / lrow[r];
        size_t row = rowbase + q0 + mbase + r;
        o[row * Cc + h * 32 + bcol]      = o0[r] * inv;
        o[row * Cc + h * 32 + 16 + bcol] = o1[r] * inv;
    }
}

// ---------------------------------------------------------------------------
// Graph / elementwise / BN kernels
// ---------------------------------------------------------------------------
__global__ void k_zero(float* __restrict__ p, long long n) {
    long long i = (long long)blockIdx.x * blockDim.x + threadIdx.x;
    long long stride = (long long)gridDim.x * blockDim.x;
    for (; i < n; i += stride) p[i] = 0.0f;
}

__global__ void k_deg(const int* __restrict__ src, float* __restrict__ deg, int E) {
    int e = blockIdx.x * blockDim.x + threadIdx.x;
    if (e < E) atomicAdd(&deg[src[e]], 1.0f);
}

__global__ void k_dis(float* __restrict__ deg, int n) {
    int i = blockIdx.x * blockDim.x + threadIdx.x;
    if (i < n) { float d = deg[i]; deg[i] = d > 0.0f ? rsqrtf(d) : 0.0f; }
}

// agg[dst] += dis[src]*dis[dst] * v[src]   (one thread = 1 edge x 4 channels)
__global__ void k_agg(const int* __restrict__ src, const int* __restrict__ dst,
                      const float* __restrict__ dis, const float* __restrict__ v,
                      float* __restrict__ agg, int E)
{
    int gid = blockIdx.x * blockDim.x + threadIdx.x;
    int e = gid >> 5;
    if (e >= E) return;
    int c4 = (gid & 31) << 2;
    int s = src[e], d = dst[e];
    float w = dis[s] * dis[d];
    const float4 vv = *(const float4*)(v + (size_t)s * Cc + c4);
    float* ap = agg + (size_t)d * Cc + c4;
    atomicAdd(ap + 0, w * vv.x);
    atomicAdd(ap + 1, w * vv.y);
    atomicAdd(ap + 2, w * vv.z);
    atomicAdd(ap + 3, w * vv.w);
}

__global__ void k_cheb_first(const float* __restrict__ v, const float* __restrict__ agg,
                             float* __restrict__ tx1, const float* __restrict__ lam, long long n)
{
    long long i = (long long)blockIdx.x * blockDim.x + threadIdx.x;
    if (i < n) {
        float scale = 2.0f / lam[0];
        tx1[i] = (scale - 1.0f) * v[i] - scale * agg[i];
    }
}

__global__ void k_cheb_next(const float* __restrict__ tx1, const float* __restrict__ agg,
                            const float* __restrict__ tx0, float* __restrict__ tx2,
                            const float* __restrict__ lam, long long n)
{
    long long i = (long long)blockIdx.x * blockDim.x + threadIdx.x;
    if (i < n) {
        float scale = 2.0f / lam[0];
        tx2[i] = 2.0f * ((scale - 1.0f) * tx1[i] - scale * agg[i]) - tx0[i];
    }
}

__global__ void k_slam(float* __restrict__ uth, const float* __restrict__ Lambda,
                       const float* __restrict__ gamma, int total, int c)
{
    int i = blockIdx.x * blockDim.x + threadIdx.x;
    if (i < total) {
        int r = i / c;
        uth[i] *= __expf(-gamma[0] * Lambda[r] * Lambda[r]);
    }
}

__global__ void k_add3(const float* __restrict__ a, const float* __restrict__ b,
                       const float* __restrict__ c, float* __restrict__ out, long long n)
{
    long long i = (long long)blockIdx.x * blockDim.x + threadIdx.x;
    if (i < n) out[i] = a[i] + b[i] + c[i];
}

__global__ void k_add2(float* __restrict__ a, const float* __restrict__ b, long long n)
{
    long long i = (long long)blockIdx.x * blockDim.x + threadIdx.x;
    if (i < n) a[i] += b[i];
}

__global__ void k_bn_stats(const float* __restrict__ h, float* __restrict__ mu,
                           float* __restrict__ var, int n, int c)
{
    __shared__ float s1[256], s2[256];
    int ch = blockIdx.x, tid = threadIdx.x;
    float a = 0.0f, b = 0.0f;
    for (int r = tid; r < n; r += 256) {
        float v = h[(size_t)r * c + ch];
        a += v; b += v * v;
    }
    s1[tid] = a; s2[tid] = b;
    __syncthreads();
    for (int s = 128; s > 0; s >>= 1) {
        if (tid < s) { s1[tid] += s1[tid + s]; s2[tid] += s2[tid + s]; }
        __syncthreads();
    }
    if (tid == 0) {
        float m = s1[0] / n;
        mu[ch] = m;
        var[ch] = s2[0] / n - m * m;
    }
}

__global__ void k_bn_apply(const float* __restrict__ h, const float* __restrict__ mu,
                           const float* __restrict__ var, const float* __restrict__ w,
                           const float* __restrict__ b, float* __restrict__ out,
                           long long total, int c)
{
    long long i = (long long)blockIdx.x * blockDim.x + threadIdx.x;
    if (i < total) {
        int ch = (int)(i % c);
        out[i] = (h[i] - mu[ch]) * rsqrtf(var[ch] + 1e-5f) * w[ch] + b[ch];
    }
}

// ---------------------------------------------------------------------------
extern "C" void kernel_launch(void* const* d_in, const int* in_sizes, int n_in,
                              void* d_out, int out_size, void* d_ws, size_t ws_size,
                              hipStream_t stream)
{
    const float* x        = (const float*)d_in[0];
    const float* U        = (const float*)d_in[1];
    const float* Lambda   = (const float*)d_in[2];
    const float* lam_max  = (const float*)d_in[3];
    const float* w_spa1   = (const float*)d_in[4];
    const float* b_spa1   = (const float*)d_in[5];
    const float* w_spa2   = (const float*)d_in[6];
    const float* b_spa2   = (const float*)d_in[7];
    const float* w_spe1   = (const float*)d_in[8];
    const float* b_spe1   = (const float*)d_in[9];
    const float* w_spe2   = (const float*)d_in[10];
    const float* b_spe2   = (const float*)d_in[11];
    const float* cheb_w   = (const float*)d_in[12];
    const float* cheb_b   = (const float*)d_in[13];
    const float* gamma    = (const float*)d_in[14];
    const float* w_proj   = (const float*)d_in[15];
    const float* w_qkv    = (const float*)d_in[16];
    const float* b_qkv    = (const float*)d_in[17];
    const float* w_out    = (const float*)d_in[18];
    const float* b_out    = (const float*)d_in[19];
    const float* mlp_w1   = (const float*)d_in[20];
    const float* mlp_b1   = (const float*)d_in[21];
    const float* mlp_w2   = (const float*)d_in[22];
    const float* mlp_b2   = (const float*)d_in[23];
    const float* bn1_w    = (const float*)d_in[24];
    const float* bn1_b    = (const float*)d_in[25];
    const float* bn2_w    = (const float*)d_in[26];
    const float* bn2_b    = (const float*)d_in[27];
    const float* bn3_w    = (const float*)d_in[28];
    const float* bn3_b    = (const float*)d_in[29];
    const int*   edge     = (const int*)d_in[30];
    const int*   src      = edge;
    const int*   dst      = edge + Ee;

    const size_t NC = (size_t)Nn * Cc;
    float* ws = (float*)d_ws;
    float* bufs[6];
    for (int i = 0; i < 6; ++i) bufs[i] = ws + (size_t)i * NC;
    float* deg = ws + 6 * NC;
    float* uth = deg + Nn;           // [KEIG, C]
    float* mu  = uth + (size_t)KEIG * Cc;
    float* var = mu + Cc;

    const long long NCll = (long long)NC;
    const int EW_BLK = 256;
    const int ew_grid = (int)((NCll + EW_BLK - 1) / EW_BLK);

    auto gemm = [&](const float* A, const float* W, const float* bias,
                    const float* resid, float* out, int M, int K, int Nc, int relu) {
        dim3 g(Nc / 64, M / 128);
        k_gemm<<<g, 256, 0, stream>>>(A, W, bias, resid, out, M, K, Nc, relu);
    };

    // ---- Phase 1: local spectral conv (Chebyshev) ----
    gemm(x, w_spa1, b_spa1, nullptr, bufs[0], Nn, Cc, Cc, 1);
    gemm(bufs[0], w_spa2, b_spa2, nullptr, bufs[1], Nn, Cc, Cc, 0);   // Tx0 = x_sp

    k_zero<<<256, 256, 0, stream>>>(deg, Nn);
    k_deg<<<Ee / 256, 256, 0, stream>>>(src, deg, Ee);
    k_dis<<<Nn / 256, 256, 0, stream>>>(deg, Nn);

    float* Tx0 = bufs[1];
    float* Tx1 = bufs[2];
    float* Tx2 = bufs[3];
    float* agg = bufs[0];
    float* outloc = bufs[4];

    gemm(Tx0, cheb_w, nullptr, nullptr, outloc, Nn, Cc, Cc, 0);       // k=0

    k_zero<<<4096, 256, 0, stream>>>(agg, NCll);
    k_agg<<<(Ee * 32) / 256, 256, 0, stream>>>(src, dst, deg, Tx0, agg, Ee);
    k_cheb_first<<<ew_grid, EW_BLK, 0, stream>>>(Tx0, agg, Tx1, lam_max, NCll);
    gemm(Tx1, cheb_w + (size_t)1 * Cc * Cc, nullptr, outloc, outloc, Nn, Cc, Cc, 0);

    for (int k = 2; k < KCH; ++k) {
        k_zero<<<4096, 256, 0, stream>>>(agg, NCll);
        k_agg<<<(Ee * 32) / 256, 256, 0, stream>>>(src, dst, deg, Tx1, agg, Ee);
        k_cheb_next<<<ew_grid, EW_BLK, 0, stream>>>(Tx1, agg, Tx0, Tx2, lam_max, NCll);
        gemm(Tx2, cheb_w + (size_t)k * Cc * Cc,
             (k == KCH - 1) ? cheb_b : nullptr, outloc, outloc, Nn, Cc, Cc, 0);
        float* t = Tx0; Tx0 = Tx1; Tx1 = Tx2; Tx2 = t;
    }

    // ---- Phase 2: global spectral branch ----
    gemm(x, w_spe1, b_spe1, nullptr, bufs[0], Nn, Cc, Cc, 1);
    gemm(bufs[0], w_spe2, b_spe2, nullptr, bufs[1], Nn, Cc, Cc, 0);
    gemm(bufs[1], w_proj, nullptr, nullptr, bufs[2], Nn, Cc, Cc, 0);  // h_proj

    k_zero<<<64, 256, 0, stream>>>(uth, (long long)KEIG * Cc);
    {
        dim3 g(Cc / 64, KEIG / 128, 64);
        k_gemm_atb<<<g, 256, 0, stream>>>(U, bufs[2], uth, KEIG, Nn, Cc, Nn / 64);
    }
    k_slam<<<(KEIG * Cc) / 256, 256, 0, stream>>>(uth, Lambda, gamma, KEIG * Cc, Cc);
    gemm(U, uth, nullptr, nullptr, bufs[3], Nn, KEIG, Cc, 0);         // out_spectral

    k_add3<<<ew_grid, EW_BLK, 0, stream>>>(bufs[4], bufs[3], x, bufs[4], NCll); // h1 pre-BN
    k_bn_stats<<<Cc, 256, 0, stream>>>(bufs[4], mu, var, Nn, Cc);
    k_bn_apply<<<ew_grid, EW_BLK, 0, stream>>>(bufs[4], mu, var, bn1_w, bn1_b, bufs[4], NCll, Cc);

    // ---- Phase 3: global attention branch ----
    gemm(x, w_qkv, b_qkv, nullptr, bufs[0], Nn, Cc, 3 * Cc, 0);       // qkv in bufs[0..2]
    {
        dim3 g(Bg * Hh, NG / 64);
        k_attn<<<g, 128, 0, stream>>>(bufs[0], bufs[3]);
    }
    gemm(bufs[3], w_out, b_out, x, bufs[5], Nn, Cc, Cc, 0);           // h2 pre-BN
    k_bn_stats<<<Cc, 256, 0, stream>>>(bufs[5], mu, var, Nn, Cc);
    k_bn_apply<<<ew_grid, EW_BLK, 0, stream>>>(bufs[5], mu, var, bn2_w, bn2_b, bufs[5], NCll, Cc);

    k_add2<<<ew_grid, EW_BLK, 0, stream>>>(bufs[4], bufs[5], NCll);   // comb = h1 + h2

    // ---- Phase 4: MLP + final BN ----
    gemm(bufs[4], mlp_w1, mlp_b1, nullptr, bufs[0], Nn, Cc, 2 * Cc, 1);   // hidden (bufs[0..1])
    gemm(bufs[0], mlp_w2, mlp_b2, bufs[4], bufs[2], Nn, 2 * Cc, Cc, 0);
    k_bn_stats<<<Cc, 256, 0, stream>>>(bufs[2], mu, var, Nn, Cc);
    k_bn_apply<<<ew_grid, EW_BLK, 0, stream>>>(bufs[2], mu, var, bn3_w, bn3_b,
                                               (float*)d_out, NCll, Cc);
}
